// MultiheadClassifier_30382598652209
// MI455X (gfx1250) — matrix operand order repaired and checked by reading, not compiled, not run on hardware
//
#include <hip/hip_runtime.h>

// ---------------- problem constants ----------------
constexpr int NB   = 4;      // batch
constexpr int NS   = 1001;   // sequence
constexpr int EIN  = 256;    // in embedding
constexpr int EOUT = 64;     // per-head out
constexpr int NH   = 10;     // heads
constexpr int SP   = 1024;   // padded sequence (64 tiles of 16, 32 k-tiles of 32)
constexpr int SCOL = 1040;   // LDS row stride in floats (1040 % 64 = 16 -> no bank aliasing)
constexpr int CONCAT_W = NH * EOUT;  // 640

typedef __attribute__((ext_vector_type(16))) __bf16 v16bf;
typedef __attribute__((ext_vector_type(8)))  float  v8f;

union AFrag {
    v16bf    v;
    unsigned u[8];
    __bf16   h[16];
};

__device__ __forceinline__ v8f wmma_bf16(v16bf a, v16bf b, v8f c) {
    // (neg_a, A, neg_b, B, c_mod, C, reuse_a, reuse_b)
    return __builtin_amdgcn_wmma_f32_16x16x32_bf16(false, a, false, b, (short)0, c,
                                                   false, false);
}

// ---------------- workspace layout (bytes) ----------------
constexpr size_t XB_BYTES = (size_t)NB * SP * EIN * 2;            // 2,097,152
constexpr size_t WB_BYTES = (size_t)3 * NH * EOUT * EIN * 2;      //   983,040
constexpr size_t QK_BYTES = (size_t)NB * NH * SP * EOUT * 2;      // 5,242,880
constexpr size_t XB_OFF = 0;
constexpr size_t WB_OFF = XB_OFF + XB_BYTES;
constexpr size_t Q_OFF  = WB_OFF + WB_BYTES;
constexpr size_t K_OFF  = Q_OFF + QK_BYTES;
constexpr size_t V_OFF  = K_OFF + QK_BYTES;   // Vt transposed [b,h,o,SP]

// ---------------- kernel 1: x -> bf16, zero-padded rows ----------------
__global__ void cvt_x_kernel(const float* __restrict__ x, __bf16* __restrict__ xb) {
    int i = blockIdx.x * 256 + threadIdx.x;          // over NB*SP*EIN = 2^20
    int e = i & (EIN - 1);
    int s = (i >> 8) & (SP - 1);
    int b = i >> 18;
    float v = (s < NS) ? x[((size_t)b * NS + s) * EIN + e] : 0.0f;
    xb[i] = (__bf16)v;
}

// ---------------- kernel 2: Wq/Wk/Wv -> bf16 (stacked) ----------------
__global__ void cvt_w_kernel(const float* __restrict__ wq, const float* __restrict__ wk,
                             const float* __restrict__ wv, __bf16* __restrict__ wb) {
    int i = blockIdx.x * 256 + threadIdx.x;          // over 3*NH*EOUT*EIN
    const int per = NH * EOUT * EIN;                 // 163840
    int mat = i / per, j = i - mat * per;
    const float* src = (mat == 0) ? wq : (mat == 1) ? wk : wv;
    wb[i] = (__bf16)src[j];
}

// ---------------- kernel 3: fused QKV projection (WMMA bf16) ----------------
// grid = (64 stiles, NH, NB), block = 256 (8 waves)
__global__ void qkv_proj_kernel(const __bf16* __restrict__ xb, const __bf16* __restrict__ wb,
                                const float* __restrict__ bq, const float* __restrict__ bk,
                                const float* __restrict__ bv,
                                __bf16* __restrict__ Q, __bf16* __restrict__ Kt,
                                __bf16* __restrict__ Vt) {
    const int stile = blockIdx.x, h = blockIdx.y, b = blockIdx.z;
    const int lane = threadIdx.x & 31, wave = threadIdx.x >> 5;
    const int s0 = stile * 16;
    const int mrow = lane & 15, half = lane >> 4;
    const size_t bh = (size_t)b * NH + h;

    const __bf16* xrow = xb + ((size_t)b * SP + s0 + mrow) * EIN;

    for (int job = wave; job < 12; job += 8) {       // wave-uniform trip counts
        const int mat = job >> 2;                    // 0=q 1=k 2=v
        const int nt  = job & 3;                     // n-tile of EOUT
        const int oc  = nt * 16 + mrow;              // this lane's output column
        const __bf16* wrow = wb + ((size_t)mat * NH + h) * EOUT * EIN
                                + (size_t)oc * EIN + half * 16;
        v8f c = {0.f, 0.f, 0.f, 0.f, 0.f, 0.f, 0.f, 0.f};
        for (int kt = 0; kt < 8; ++kt) {             // K = 256 in steps of 32
            AFrag a;
            const __bf16* ar = xrow + kt * 32;
#pragma unroll
            for (int p = 0; p < 8; ++p) {            // contiguous A layout k = 16*half + s
                int k0 = half * 16 + 2 * p;
                a.u[p] = *(const unsigned*)(ar + k0);
            }
            v16bf bf = *(const v16bf*)(wrow + kt * 32);   // contiguous 32B B-frag
            c = wmma_bf16(a.v, bf, c);
        }
        const float* bias = (mat == 0) ? bq : (mat == 1) ? bk : bv;
        const float bo = bias[h * EOUT + oc];
        if (mat < 2) {
            __bf16* dst = ((mat == 0) ? Q : Kt) + (bh * SP + s0) * EOUT + oc;
#pragma unroll
            for (int r = 0; r < 8; ++r) {
                int m = r + half * 8;
                dst[(size_t)m * EOUT] = (__bf16)(c[r] + bo);
            }
        } else {                                     // V stored transposed [o][t]
            __bf16* dst = Vt + (bh * EOUT + oc) * (size_t)SP + s0 + half * 8;
#pragma unroll
            for (int r = 0; r < 8; ++r) dst[r] = (__bf16)(c[r] + bo);
        }
    }
}

// ---------------- kernel 4: fused attention ----------------
// grid = (63 row-tiles, NB), block = 256 (8 waves); loops heads internally
// dynamic LDS: scores[16][SCOL] f32  +  partials[8][16][64] f32
__global__ void attn_kernel(const __bf16* __restrict__ Q, const __bf16* __restrict__ Kt,
                            const __bf16* __restrict__ Vt, const float* __restrict__ lam,
                            float* __restrict__ outC, float* __restrict__ outW) {
    extern __shared__ float smem[];
    float* sc   = smem;                 // 16*SCOL
    float* part = smem + 16 * SCOL;     // 8*16*64

    const int stile = blockIdx.x, b = blockIdx.y;
    const int lane = threadIdx.x & 31, wave = threadIdx.x >> 5;
    const int s0 = stile * 16;
    const int mrow = lane & 15, half = lane >> 4;

    for (int h = 0; h < NH; ++h) {
        const size_t bh = (size_t)b * NH + h;

        // -------- load this tile's Q A-fragments (reused for all 64 column tiles)
        const __bf16* qr = Q + (bh * SP + s0 + mrow) * EOUT;
        AFrag a0, a1;
#pragma unroll
        for (int p = 0; p < 8; ++p) {
            int k0 = half * 16 + 2 * p;
            a0.u[p] = *(const unsigned*)(qr + k0);
            a1.u[p] = *(const unsigned*)(qr + 32 + k0);
        }

        // -------- step 1: scores = Q*K^T / 16 into LDS
        for (int ct = wave; ct < 64; ct += 8) {      // 8 tiles per wave, uniform
            const int t0 = ct * 16;
            const __bf16* kr = Kt + (bh * SP + t0 + mrow) * EOUT + half * 16;
            __builtin_prefetch(kr + 128 * EOUT, 0, 1);   // next owned tile
            v16bf b0 = *(const v16bf*)(kr);
            v16bf b1 = *(const v16bf*)(kr + 32);
            v8f c = {0.f, 0.f, 0.f, 0.f, 0.f, 0.f, 0.f, 0.f};
            c = wmma_bf16(a0.v, b0, c);
            c = wmma_bf16(a1.v, b1, c);
#pragma unroll
            for (int r = 0; r < 8; ++r)
                sc[(r + half * 8) * SCOL + t0 + mrow] = c[r] * 0.0625f;  // 1/sqrt(256)
        }
        __syncthreads();

        // -------- step 2: softmax -> *mask -> L2 normalize -> weights_sum accum
        for (int rr = 0; rr < 2; ++rr) {             // 2 rows per wave
            const int row = wave * 2 + rr;
            const int s = s0 + row;
            const bool valid = s < NS;               // wave-uniform
            float* r_ = sc + row * SCOL;

            float mx = -3.4e38f;
            for (int t = lane; t < NS; t += 32) mx = fmaxf(mx, r_[t]);
            for (int off = 16; off; off >>= 1) mx = fmaxf(mx, __shfl_xor(mx, off, 32));

            float sum = 0.f;
            for (int t = lane; t < NS; t += 32) {
                float e = __expf(r_[t] - mx);
                r_[t] = e; sum += e;
            }
            for (int off = 16; off; off >>= 1) sum += __shfl_xor(sum, off, 32);
            const float inv = 1.f / sum;

            const float* lrow = lam + (size_t)(valid ? s : 0) * NS;
            float ss = 0.f;
            for (int t = lane; t < NS; t += 32) {
                float loc = r_[t] * inv * lrow[t];
                r_[t] = loc; ss += loc * loc;
            }
            for (int off = 16; off; off >>= 1) ss += __shfl_xor(ss, off, 32);
            const float rdn = 1.f / fmaxf(__builtin_sqrtf(ss), 1e-12f);

            float* wrow = outW + ((size_t)b * NS + (valid ? s : 0)) * NS;
            for (int t = lane; t < NS; t += 32) {
                float w = r_[t] * rdn;
                r_[t] = w;
                if (valid) {                          // WG exclusively owns these rows
                    if (h == 0) wrow[t] = w;
                    else        wrow[t] += w;
                }
            }
            for (int t = NS + lane; t < SP; t += 32) r_[t] = 0.f;   // zero padding
        }
        __syncthreads();

        // -------- step 3: out = w * V  (K = SP in 32 tiles of 32)
        v8f c0 = {0.f,0.f,0.f,0.f,0.f,0.f,0.f,0.f};
        v8f c1 = c0, c2 = c0, c3 = c0;
        const __bf16* vbb = Vt + bh * (size_t)EOUT * SP + (size_t)mrow * SP + half * 16;
        for (int kt = wave; kt < 32; kt += 8) {      // 4 tiles per wave, uniform
            AFrag a;
#pragma unroll
            for (int p = 0; p < 8; ++p) {
                int k = kt * 32 + half * 16 + 2 * p;
                float2 f = *(const float2*)&sc[mrow * SCOL + k];
                a.h[2 * p]     = (__bf16)f.x;
                a.h[2 * p + 1] = (__bf16)f.y;
            }
            const __bf16* vb = vbb + kt * 32;
            c0 = wmma_bf16(a.v, *(const v16bf*)(vb + 0 * 16 * SP), c0);
            c1 = wmma_bf16(a.v, *(const v16bf*)(vb + 1 * 16 * SP), c1);
            c2 = wmma_bf16(a.v, *(const v16bf*)(vb + 2 * 16 * SP), c2);
            c3 = wmma_bf16(a.v, *(const v16bf*)(vb + 3 * 16 * SP), c3);
        }
        float* pw = part + wave * (16 * 64);
#pragma unroll
        for (int r = 0; r < 8; ++r) {
            int m = r + half * 8;
            pw[m * 64 +  0 + mrow] = c0[r];
            pw[m * 64 + 16 + mrow] = c1[r];
            pw[m * 64 + 32 + mrow] = c2[r];
            pw[m * 64 + 48 + mrow] = c3[r];
        }
        __syncthreads();

        // -------- fixed-order cross-wave reduction (deterministic) + writeout
        for (int j = 0; j < 4; ++j) {
            int idx = threadIdx.x + j * 256;         // = m*64 + o
            int m = idx >> 6, o = idx & 63;
            float acc = 0.f;
#pragma unroll
            for (int wv = 0; wv < 8; ++wv) acc += part[wv * 1024 + idx];
            if (s0 + m < NS)
                outC[((size_t)b * NS + s0 + m) * CONCAT_W + h * EOUT + o] = acc;
        }
        __syncthreads();
    }
}

// ---------------- launcher ----------------
extern "C" void kernel_launch(void* const* d_in, const int* in_sizes, int n_in,
                              void* d_out, int out_size, void* d_ws, size_t ws_size,
                              hipStream_t stream) {
    const float* x   = (const float*)d_in[0];
    const float* lam = (const float*)d_in[1];
    const float* Wq  = (const float*)d_in[2];
    const float* bq  = (const float*)d_in[3];
    const float* Wk  = (const float*)d_in[4];
    const float* bk  = (const float*)d_in[5];
    const float* Wv  = (const float*)d_in[6];
    const float* bv  = (const float*)d_in[7];

    float* outC = (float*)d_out;                               // [B,S,640]
    float* outW = outC + (size_t)NB * NS * CONCAT_W;           // [B,S,S]

    char* ws = (char*)d_ws;
    __bf16* xb = (__bf16*)(ws + XB_OFF);
    __bf16* wb = (__bf16*)(ws + WB_OFF);
    __bf16* Qw = (__bf16*)(ws + Q_OFF);
    __bf16* Kw = (__bf16*)(ws + K_OFF);
    __bf16* Vt = (__bf16*)(ws + V_OFF);

    cvt_x_kernel<<<dim3((NB * SP * EIN) / 256), dim3(256), 0, stream>>>(x, xb);
    cvt_w_kernel<<<dim3((3 * NH * EOUT * EIN) / 256), dim3(256), 0, stream>>>(Wq, Wk, Wv, wb);
    qkv_proj_kernel<<<dim3(SP / 16, NH, NB), dim3(256), 0, stream>>>(xb, wb, bq, bk, bv,
                                                                     Qw, Kw, Vt);
    const size_t smem = (size_t)(16 * SCOL + 8 * 16 * 64) * sizeof(float);  // ~97 KB
    attn_kernel<<<dim3((NS + 15) / 16, NB), dim3(256), smem, stream>>>(Qw, Kw, Vt, lam,
                                                                       outC, outW);
}
